// GNNEncoder_14388140441811
// MI455X (gfx1250) — matrix operand order repaired and checked by reading, not compiled, not run on hardware
//
#include <hip/hip_runtime.h>
#include <stdint.h>
#include <stddef.h>

typedef __attribute__((ext_vector_type(16))) __bf16 v16bf;
typedef __attribute__((ext_vector_type(8)))  __bf16 v8bf;
typedef __attribute__((ext_vector_type(2)))  __bf16 v2bf;
typedef __attribute__((ext_vector_type(8)))  float  v8f;

#define IN_CH  128
#define HID_CH 128
#define OUT_CH 64

// f32 -> bf16 (RNE). Native cast lowers to hardware cvt on gfx1250.
__device__ __forceinline__ __bf16 f2bf(float f) { return (__bf16)f; }

// Packed pair conversion: 1 VALU op for 2 elements when the builtin exists.
__device__ __forceinline__ v2bf pk2(float lo, float hi) {
#if __has_builtin(__builtin_amdgcn_cvt_pk_bf16_f32)
    return __builtin_amdgcn_cvt_pk_bf16_f32(lo, hi);
#else
    v2bf r; r[0] = (__bf16)lo; r[1] = (__bf16)hi; return r;
#endif
}

__device__ __forceinline__ void atomAdd(float* p, float v) {
    __hip_atomic_fetch_add(p, v, __ATOMIC_RELAXED, __HIP_MEMORY_SCOPE_AGENT);
}

// ---------------- degree / normalization ----------------
__global__ void k_deg_init(float* deg, int n) {
    int i = blockIdx.x * blockDim.x + threadIdx.x;
    if (i < n) deg[i] = 1.0f;  // self-loop contributes 1
}

__global__ void k_deg_accum(const long long* __restrict__ ei, float* __restrict__ deg, int E) {
    int e = blockIdx.x * blockDim.x + threadIdx.x;
    if (e < E) atomAdd(&deg[(int)ei[(long long)E + e]], 1.0f);
}

__global__ void k_dinv(float* deg, int n) {
    int i = blockIdx.x * blockDim.x + threadIdx.x;
    if (i < n) deg[i] = rsqrtf(deg[i]);  // deg >= 1 always (self-loop)
}

// ---------------- WMMA GEMM: C[nrows x NOUT] = A[nrows x 128] * W[128 x NOUT] ----------------
// 256 threads = 8 waves; each wave computes a 16-row strip across all NOUT columns.
// W is staged in LDS transposed [n][k] so each lane's B fragment is two contiguous runs of 8 bf16 (16B each).
template<int NOUT, bool A_IS_BF16>
__global__ void k_gemm(const void* __restrict__ Aptr, const float* __restrict__ W,
                       float* __restrict__ C, int nrows) {
    __shared__ __bf16 Ws[NOUT * 128];  // transposed: Ws[n*128 + k] = W[k][n]

    int tid = threadIdx.x;
    for (int i = tid; i < 128 * NOUT; i += 256) {
        int k = i / NOUT, n = i % NOUT;          // linear over W -> coalesced global read
        Ws[n * 128 + k] = f2bf(W[i]);
    }
    __syncthreads();

    int wave = tid >> 5, lane = tid & 31;
    int half = lane >> 4, lm = lane & 15;
    long long strip = (long long)blockIdx.x * 8 + wave;
    if (strip * 16 >= nrows) return;
    int row0 = (int)(strip * 16);
    int m = row0 + lm;
    if (m >= nrows) m = nrows - 1;               // clamp loads; stores guarded below

    constexpr int NT = NOUT / 16;
    v8f acc[NT] = {};

    #pragma unroll
    for (int kc = 0; kc < 4; ++kc) {             // K = 128 in chunks of 32
        int kbase = kc * 32;
        // A fragment (16x32 bf16): lane's 16 elems are two contiguous 8-element runs:
        //   a[0..7]  <- K = kbase +      half*8 .. +8
        //   a[8..15] <- K = kbase + 16 + half*8 .. +8
        v16bf a;
        if constexpr (A_IS_BF16) {
            const __bf16* ar = (const __bf16*)Aptr + (long long)m * 128;
            v8bf q0 = *(const v8bf*)(ar + kbase + half * 8);
            v8bf q1 = *(const v8bf*)(ar + kbase + 16 + half * 8);
            #pragma unroll
            for (int i = 0; i < 8; ++i) { a[i] = q0[i]; a[8 + i] = q1[i]; }
        } else {
            const float* ar = (const float*)Aptr + (long long)m * 128;
            const float* p0 = ar + kbase + half * 8;
            const float* p1 = ar + kbase + 16 + half * 8;
            float4 q0 = *(const float4*)(p0);
            float4 q1 = *(const float4*)(p0 + 4);
            float4 q2 = *(const float4*)(p1);
            float4 q3 = *(const float4*)(p1 + 4);
            v2bf c0 = pk2(q0.x, q0.y), c1 = pk2(q0.z, q0.w);
            v2bf c2 = pk2(q1.x, q1.y), c3 = pk2(q1.z, q1.w);
            v2bf c4 = pk2(q2.x, q2.y), c5 = pk2(q2.z, q2.w);
            v2bf c6 = pk2(q3.x, q3.y), c7 = pk2(q3.z, q3.w);
            a[0]  = c0[0]; a[1]  = c0[1]; a[2]  = c1[0]; a[3]  = c1[1];
            a[4]  = c2[0]; a[5]  = c2[1]; a[6]  = c3[0]; a[7]  = c3[1];
            a[8]  = c4[0]; a[9]  = c4[1]; a[10] = c5[0]; a[11] = c5[1];
            a[12] = c6[0]; a[13] = c6[1]; a[14] = c7[0]; a[15] = c7[1];
        }
        #pragma unroll
        for (int t = 0; t < NT; ++t) {
            // B fragment (32x16 bf16): b[0..7] = Ws[n][kbase + half*8 .. +8], b[8..15] = Ws[n][kbase + 16 + half*8 .. +8]
            int n = t * 16 + lm;
            const __bf16* wp = &Ws[n * 128 + kbase + half * 8];
            v8bf r0 = *(const v8bf*)wp;
            v8bf r1 = *(const v8bf*)(wp + 16);
            v16bf b;
            #pragma unroll
            for (int i = 0; i < 8; ++i) { b[i] = r0[i]; b[8 + i] = r1[i]; }
            acc[t] = __builtin_amdgcn_wmma_f32_16x16x32_bf16(
                false, a, false, b, (short)0, acc[t], false, false);
        }
    }

    // C/D layout: VGPR v -> row (v + 8*half), column = t*16 + lm
    #pragma unroll
    for (int t = 0; t < NT; ++t) {
        int n = t * 16 + lm;
        #pragma unroll
        for (int v = 0; v < 8; ++v) {
            int mr = row0 + v + 8 * half;
            if (mr < nrows) C[(long long)mr * NOUT + n] = acc[t][v];
        }
    }
}

// ---------------- self-loop init (+ optional bias): out[i][c] = h[i][c]*dinv[i]^2 (+ b[c]) ----
template<int CH, int ADD_BIAS>
__global__ void k_self_init(const float* __restrict__ h, const float* __restrict__ dinv,
                            const float* __restrict__ bias, float* __restrict__ out,
                            long long total) {
    long long i = (long long)blockIdx.x * blockDim.x + threadIdx.x;
    if (i >= total) return;
    int node = (int)(i / CH), c = (int)(i & (CH - 1));   // CH is a power of two
    float w = dinv[node]; w *= w;
    float v = h[i] * w;
    if (ADD_BIAS) v += bias[c];
    out[i] = v;
}

// ---------------- edge scatter: one wave per edge, coalesced 512B/256B per wave --------------
template<int C>
__global__ void k_edge_scatter(const float* __restrict__ h, const long long* __restrict__ ei,
                               const float* __restrict__ dinv, float* __restrict__ out, int E) {
    long long t = (long long)blockIdx.x * blockDim.x + threadIdx.x;
    int e = (int)(t >> 5), lane = (int)(t & 31);
    if (e >= E) return;
    int s = (int)ei[e];
    int d = (int)ei[(long long)E + e];
    float w = dinv[s] * dinv[d];
    if constexpr (C == 128) {
        float4 v = ((const float4*)(h + (long long)s * 128))[lane];
        float* o = out + (long long)d * 128 + lane * 4;
        atomAdd(o + 0, v.x * w); atomAdd(o + 1, v.y * w);
        atomAdd(o + 2, v.z * w); atomAdd(o + 3, v.w * w);
    } else {
        float2 v = ((const float2*)(h + (long long)s * 64))[lane];
        float* o = out + (long long)d * 64 + lane * 2;
        atomAdd(o + 0, v.x * w); atomAdd(o + 1, v.y * w);
    }
}

// ---------------- bias + ReLU + bf16 pack for layer-2 GEMM input ----------------
__global__ void k_relu_bf16(const float* __restrict__ agg, const float* __restrict__ b1,
                            __bf16* __restrict__ hb, long long total) {
    long long i = (long long)blockIdx.x * blockDim.x + threadIdx.x;
    if (i >= total) return;
    int c = (int)(i & 127);
    float v = agg[i] + b1[c];
    hb[i] = f2bf(v > 0.f ? v : 0.f);
}

static inline int ceil_div_i(long long a, long long b) { return (int)((a + b - 1) / b); }

extern "C" void kernel_launch(void* const* d_in, const int* in_sizes, int n_in,
                              void* d_out, int out_size, void* d_ws, size_t ws_size,
                              hipStream_t stream) {
    const float*     x  = (const float*)d_in[0];
    const long long* ei = (const long long*)d_in[1];   // int64, shape [2, E]
    const float*     W1 = (const float*)d_in[2];
    const float*     b1 = (const float*)d_in[3];
    const float*     W2 = (const float*)d_in[4];
    const float*     b2 = (const float*)d_in[5];
    float* out = (float*)d_out;

    int n = in_sizes[0] / IN_CH;
    int E = in_sizes[1] / 2;

    // Workspace layout (aliased, ~103 MB; all live data fits MI455X's 192MB L2):
    //   dinv : n f32
    //   h1   : n*128 f32   (later: low half = h1b bf16, high half = h2 f32 — disjoint lifetimes)
    //   agg1 : n*128 f32
    char* ws = (char*)d_ws;
    float*  dinv = (float*)ws;
    size_t  off1 = ((size_t)n * 4 + 255) & ~(size_t)255;
    float*  h1   = (float*)(ws + off1);
    size_t  off2 = off1 + (size_t)n * 128 * 4;
    float*  agg1 = (float*)(ws + off2);
    __bf16* h1b  = (__bf16*)h1;                               // n*128 bf16, aliases dead h1
    float*  h2   = (float*)(ws + off1 + (size_t)n * 128 * 2); // n*64 f32, upper half of h1 region

    const int BT = 256;
    long long tot128 = (long long)n * 128;
    long long tot64  = (long long)n * 64;
    int nstrips = (n + 15) / 16;
    int gemm_blocks = (nstrips + 7) / 8;

    // ---- normalization ----
    k_deg_init <<<ceil_div_i(n, BT), BT, 0, stream>>>(dinv, n);
    k_deg_accum<<<ceil_div_i(E, BT), BT, 0, stream>>>(ei, dinv, E);
    k_dinv     <<<ceil_div_i(n, BT), BT, 0, stream>>>(dinv, n);

    // ---- layer 1: h1 = x @ W1 (bf16 WMMA, f32 accumulate) ----
    k_gemm<HID_CH, false><<<gemm_blocks, BT, 0, stream>>>((const void*)x, W1, h1, n);
    // agg1 = self-loop term; then scatter edges (L2-resident atomics)
    k_self_init<128, 0><<<ceil_div_i(tot128, BT), BT, 0, stream>>>(h1, dinv, b1, agg1, tot128);
    k_edge_scatter<128><<<ceil_div_i((long long)E * 32, BT), BT, 0, stream>>>(h1, ei, dinv, agg1, E);
    // h1b = bf16(relu(agg1 + b1))   (h1 region is dead by now)
    k_relu_bf16<<<ceil_div_i(tot128, BT), BT, 0, stream>>>(agg1, b1, h1b, tot128);

    // ---- layer 2: h2 = h1b @ W2 ----
    k_gemm<OUT_CH, true><<<gemm_blocks, BT, 0, stream>>>((const void*)h1b, W2, h2, n);
    // d_out = self-loop term + b2; then scatter edges
    k_self_init<64, 1><<<ceil_div_i(tot64, BT), BT, 0, stream>>>(h2, dinv, b2, out, tot64);
    k_edge_scatter<64><<<ceil_div_i((long long)E * 32, BT), BT, 0, stream>>>(h2, ei, dinv, out, E);
}
